// ResidualVectorQuantizer_33139967656628
// MI455X (gfx1250) — matrix operand order repaired and checked by reading, not compiled, not run on hardware
//
#include <hip/hip_runtime.h>
#include <hip/hip_bf16.h>
#include <stdint.h>

// ---------------------------------------------------------------------------
// Residual VQ for MI455X (gfx1250, wave32).
//
// Shapes: x[16,2048,256] f32, codebooks[4,1024,256] f32.
// d_out (f32, concatenated): quantized[16*2048*256], q_loss[1], e_loss[1],
//                            indices[4*16*2048] (float-coded ints).
//
// Roofline: 4 levels * 2*N*K*D ~= 69 GFLOP vs ~70MB HBM traffic (~3us at
// 23.3TB/s) -> matrix-core bound. Scores run on v_wmma_f32_16x16x32_bf16
// (f32 accum). v2: codebook tiles are staged once per 8-wave block into LDS
// (8x less L2 traffic than per-wave streaming) with a register double-buffer
// hiding the global loads behind WMMAs; two independent accumulator chains
// keep the XDL pipe busy. Gather/loss/residual update use exact f32.
// ---------------------------------------------------------------------------

#define NUM_LEVELS 4
#define NUM_CODES  1024
#define EMBED_DIM  256
#define N_VEC      (16 * 2048)          // 32768 rows
#define QN         ((size_t)N_VEC * EMBED_DIM)

#define ROWS_PER_BLOCK 128
#define LVL_THREADS    256              // 8 wave32 waves
#define LDS_STRIDE     264              // 256 + 8 bf16 pad: 528B rows -> no bank conflicts
#define KCHUNK         64               // codes staged in LDS per step
#define A_TILE_BYTES   (ROWS_PER_BLOCK * LDS_STRIDE * 2)   // 67584 (>= B stage 33792)

typedef __attribute__((ext_vector_type(16))) __bf16       v16bf;
typedef __attribute__((ext_vector_type(8)))  float        v8f;
typedef __attribute__((ext_vector_type(8)))  unsigned int v8u;
typedef __attribute__((ext_vector_type(4)))  unsigned int v4u;

static __device__ __forceinline__ __bf16 f2bf(float f) {
  unsigned int u = __builtin_bit_cast(unsigned int, f);
  unsigned int r = (u + 0x7FFFu + ((u >> 16) & 1u)) >> 16;   // round-nearest-even
  unsigned short h = (unsigned short)r;
  return __builtin_bit_cast(__bf16, h);
}

static __device__ __forceinline__ v16bf combine16(v4u lo, v4u hi) {
  v8u c;
  #pragma unroll
  for (int i = 0; i < 4; ++i) { c[i] = lo[i]; c[i + 4] = hi[i]; }
  return __builtin_bit_cast(v16bf, c);
}

// ---------------------------------------------------------------------------
// residual = x ; quant_out = 0 ; loss_acc = 0          (one float4 per thread)
// ---------------------------------------------------------------------------
__global__ void rvq_init(const float* __restrict__ x, float* __restrict__ residual,
                         float* __restrict__ quant_out, float* __restrict__ loss_acc) {
  size_t i = (size_t)blockIdx.x * blockDim.x + threadIdx.x;
  size_t n4 = QN / 4;
  if (i < n4) {
    float4 v = ((const float4*)x)[i];
    ((float4*)residual)[i] = v;
    float4 z = make_float4(0.f, 0.f, 0.f, 0.f);
    ((float4*)quant_out)[i] = z;
  }
  if (i == 0) loss_acc[0] = 0.f;
}

// ---------------------------------------------------------------------------
// Per code row (grid = L*K): bf16 copy of codebook + ||c||^2.
// ---------------------------------------------------------------------------
__global__ void rvq_cb_prep(const float* __restrict__ cb, __bf16* __restrict__ cb16,
                            float* __restrict__ cnorm) {
  __shared__ float red[EMBED_DIM];
  int row = blockIdx.x;
  int t   = threadIdx.x;
  float v = cb[(size_t)row * EMBED_DIM + t];
  cb16[(size_t)row * EMBED_DIM + t] = f2bf(v);
  red[t] = v * v;
  __syncthreads();
  for (int s = EMBED_DIM / 2; s > 0; s >>= 1) {
    if (t < s) red[t] += red[t + s];
    __syncthreads();
  }
  if (t == 0) cnorm[row] = red[0];
}

// ---------------------------------------------------------------------------
// One VQ level. Block = 256 threads (8 waves) owns 128 rows.
//   score[n,k] = ||c_k||^2 - 2 * (r_n . c_k)   (||r||^2 row-constant: dropped)
// ---------------------------------------------------------------------------
__global__ __launch_bounds__(LVL_THREADS)
void rvq_level(float* __restrict__ residual,            // [N,256] in/out (f32)
               const __bf16* __restrict__ cb16,         // [1024,256] this level
               const float*  __restrict__ cbf,          // [1024,256] this level f32
               const float*  __restrict__ cnorm,        // [1024]
               float* __restrict__ quant_acc,           // [N,256] accumulated
               float* __restrict__ idx_out,             // [N] (float-coded ints)
               float* __restrict__ loss_acc)            // [1]
{
  extern __shared__ __align__(16) char smem[];          // A tile, then B stage
  __bf16* aTile  = (__bf16*)smem;                       // [128][264]
  __bf16* bStage = (__bf16*)smem;                       // [64][264] (reused)

  __shared__ float redS[8][16][16];
  __shared__ int   redI[8][16][16];
  __shared__ int   rowIdx[ROWS_PER_BLOCK];
  __shared__ float lossRed[LVL_THREADS];

  const int tid  = threadIdx.x;
  const int row0 = blockIdx.x * ROWS_PER_BLOCK;

  // ---- Phase 1: stage residual rows into LDS as bf16 ----
  for (int e = tid; e < ROWS_PER_BLOCK * (EMBED_DIM / 4); e += LVL_THREADS) {
    int r  = e >> 6;            // 64 float4 per row
    int c4 = (e & 63) * 4;
    float4 v = *(const float4*)(residual + (size_t)(row0 + r) * EMBED_DIM + c4);
    __bf16* dst = &aTile[r * LDS_STRIDE + c4];
    dst[0] = f2bf(v.x); dst[1] = f2bf(v.y); dst[2] = f2bf(v.z); dst[3] = f2bf(v.w);
  }
  __syncthreads();

  const int wave  = tid >> 5;          // wave32
  const int lane  = tid & 31;
  const int m     = lane & 15;         // A: row M / B,D: column N
  const int khalf = lane >> 4;         // 16-bit operand K-half select
  const int wrow0 = wave * 16;         // this wave's 16 rows within block

  // ---- Preload A fragments (16 rows x 256): reused for all 64 K-tiles ----
  // ISA 16-bit A 16x32 layout: lanes0-15 K=0..7,16..23 ; lanes16-31 K=8..15,24..31
  v16bf afrag[8];
  {
    const __bf16* arow = &aTile[(wrow0 + m) * LDS_STRIDE];
    #pragma unroll
    for (int c = 0; c < 8; ++c) {
      const int d0 = c * 32;
      v4u lo = *(const v4u*)(arow + d0 + khalf * 8);
      v4u hi = *(const v4u*)(arow + d0 + 16 + khalf * 8);
      afrag[c] = combine16(lo, hi);
    }
  }

  // ---- Phase 2: scan K=1024 in 64-code LDS-staged chunks ----
  float best[8];
  int   bidx[8];
  #pragma unroll
  for (int v = 0; v < 8; ++v) { best[v] = 3.4e38f; bidx[v] = 0; }

  // Register double-buffer for the next chunk (2048 v4u per chunk / 256 thr).
  v4u breg[8];
  #pragma unroll
  for (int i = 0; i < 8; ++i) {
    int j = i * LVL_THREADS + tid;
    breg[i] = *(const v4u*)(cb16 + (size_t)(j >> 5) * EMBED_DIM + (j & 31) * 8);
  }

  for (int ch = 0; ch < NUM_CODES / KCHUNK; ++ch) {
    __syncthreads();                       // previous chunk fully consumed
    #pragma unroll
    for (int i = 0; i < 8; ++i) {          // regs -> LDS (padded rows)
      int j = i * LVL_THREADS + tid;
      *(v4u*)(bStage + (j >> 5) * LDS_STRIDE + (j & 31) * 8) = breg[i];
    }
    __syncthreads();
    if (ch + 1 < NUM_CODES / KCHUNK) {     // prefetch next chunk under WMMAs
      const __bf16* gsrc = cb16 + (size_t)(ch + 1) * KCHUNK * EMBED_DIM;
      #pragma unroll
      for (int i = 0; i < 8; ++i) {
        int j = i * LVL_THREADS + tid;
        breg[i] = *(const v4u*)(gsrc + (size_t)(j >> 5) * EMBED_DIM + (j & 31) * 8);
      }
    }

    // 4 K-tiles of 16 codes; process as 2 independent accumulator chains.
    #pragma unroll
    for (int ktp = 0; ktp < 2; ++ktp) {
      const int ktl0 = ktp * 2, ktl1 = ktp * 2 + 1;
      const int col0 = ch * KCHUNK + ktl0 * 16 + m;
      const int col1 = col0 + 16;
      const __bf16* b0 = bStage + (ktl0 * 16 + m) * LDS_STRIDE + khalf * 8;
      const __bf16* b1 = bStage + (ktl1 * 16 + m) * LDS_STRIDE + khalf * 8;
      v8f acc0 = {}, acc1 = {};
      #pragma unroll
      for (int c = 0; c < 8; ++c) {
        v16bf bf0 = combine16(*(const v4u*)(b0 + c * 32), *(const v4u*)(b0 + c * 32 + 16));
        v16bf bf1 = combine16(*(const v4u*)(b1 + c * 32), *(const v4u*)(b1 + c * 32 + 16));
        acc0 = __builtin_amdgcn_wmma_f32_16x16x32_bf16(false, afrag[c], false, bf0,
                                                       (short)0, acc0, false, false);
        acc1 = __builtin_amdgcn_wmma_f32_16x16x32_bf16(false, afrag[c], false, bf1,
                                                       (short)0, acc1, false, false);
      }
      const float cn0 = cnorm[col0];
      const float cn1 = cnorm[col1];
      #pragma unroll
      for (int v = 0; v < 8; ++v) {        // D layout: row = v + 8*khalf, col = m
        float s0 = cn0 - 2.0f * acc0[v];
        float s1 = cn1 - 2.0f * acc1[v];
        if (s0 < best[v]) { best[v] = s0; bidx[v] = col0; }
        if (s1 < best[v]) { best[v] = s1; bidx[v] = col1; }
      }
    }
  }

  // ---- Cross-lane argmin (16 lanes share each row) ----
  #pragma unroll
  for (int v = 0; v < 8; ++v) {
    redS[wave][v + 8 * khalf][m] = best[v];
    redI[wave][v + 8 * khalf][m] = bidx[v];
  }
  __syncthreads();
  if (lane < 16) {
    float bs = redS[wave][lane][0];
    int   bi = redI[wave][lane][0];
    for (int c = 1; c < 16; ++c) {
      float s  = redS[wave][lane][c];
      int   i2 = redI[wave][lane][c];
      if (s < bs || (s == bs && i2 < bi)) { bs = s; bi = i2; }
    }
    rowIdx[wrow0 + lane] = bi;
    idx_out[row0 + wrow0 + lane] = (float)bi;
  }
  __syncthreads();

  // ---- Exact f32 gather: quant+=, loss, residual-= ----
  float lsum = 0.f;
  for (int e = tid; e < ROWS_PER_BLOCK * (EMBED_DIM / 4); e += LVL_THREADS) {
    int r  = e >> 6;
    int c4 = (e & 63) * 4;
    int ci = rowIdx[r];
    size_t go = (size_t)(row0 + r) * EMBED_DIM + c4;
    float4 q   = *(const float4*)(cbf + (size_t)ci * EMBED_DIM + c4);
    float4 res = *(const float4*)(residual + go);
    float dx = q.x - res.x, dy = q.y - res.y, dz = q.z - res.z, dw = q.w - res.w;
    lsum += dx * dx + dy * dy + dz * dz + dw * dw;
    float4 qa = *(const float4*)(quant_acc + go);
    qa.x += q.x; qa.y += q.y; qa.z += q.z; qa.w += q.w;
    *(float4*)(quant_acc + go) = qa;
    res.x -= q.x; res.y -= q.y; res.z -= q.z; res.w -= q.w;
    *(float4*)(residual + go) = res;
  }
  lossRed[tid] = lsum;
  __syncthreads();
  for (int s = LVL_THREADS / 2; s > 0; s >>= 1) {
    if (tid < s) lossRed[tid] += lossRed[tid + s];
    __syncthreads();
  }
  if (tid == 0) atomicAdd(loss_acc, lossRed[0]);
}

// ---------------------------------------------------------------------------
// q_loss = e_loss = total_sq_err / (L * N * D)
// ---------------------------------------------------------------------------
__global__ void rvq_finalize(const float* __restrict__ loss_acc,
                             float* __restrict__ q_loss, float* __restrict__ e_loss) {
  if (threadIdx.x == 0 && blockIdx.x == 0) {
    float v = loss_acc[0] / ((float)NUM_LEVELS * (float)N_VEC * (float)EMBED_DIM);
    q_loss[0] = v;
    e_loss[0] = v;
  }
}

extern "C" void kernel_launch(void* const* d_in, const int* in_sizes, int n_in,
                              void* d_out, int out_size, void* d_ws, size_t ws_size,
                              hipStream_t stream) {
  (void)in_sizes; (void)n_in; (void)out_size; (void)ws_size;
  const float* x  = (const float*)d_in[0];                  // [16,2048,256]
  const float* cb = (const float*)d_in[1];                  // [4,1024,256]
  float* out = (float*)d_out;

  // Workspace: residual f32 (32MB) | cb bf16 (2MB) | cnorm (16KB) | loss (4B)
  char*   ws       = (char*)d_ws;
  float*  residual = (float*)ws;
  size_t  off      = QN * sizeof(float);
  __bf16* cb16     = (__bf16*)(ws + off);
  off += (size_t)NUM_LEVELS * NUM_CODES * EMBED_DIM * sizeof(__bf16);
  float*  cnorm    = (float*)(ws + off);
  float*  lacc     = cnorm + NUM_LEVELS * NUM_CODES;

  float* quant  = out;                 // [N,256]
  float* qloss  = out + QN;            // [1]
  float* eloss  = out + QN + 1;        // [1]
  float* idxs   = out + QN + 2;        // [L,N] float-coded

  {
    size_t n4 = QN / 4;
    int blocks = (int)((n4 + 255) / 256);
    rvq_init<<<blocks, 256, 0, stream>>>(x, residual, quant, lacc);
  }
  rvq_cb_prep<<<NUM_LEVELS * NUM_CODES, EMBED_DIM, 0, stream>>>(cb, cb16, cnorm);

  for (int l = 0; l < NUM_LEVELS; ++l) {
    rvq_level<<<N_VEC / ROWS_PER_BLOCK, LVL_THREADS, A_TILE_BYTES, stream>>>(
        residual,
        cb16  + (size_t)l * NUM_CODES * EMBED_DIM,
        cb    + (size_t)l * NUM_CODES * EMBED_DIM,
        cnorm + (size_t)l * NUM_CODES,
        quant,
        idxs + (size_t)l * N_VEC,
        lacc);
  }
  rvq_finalize<<<1, 32, 0, stream>>>(lacc, qloss, eloss);
}
